// Tlearner_23313082482708
// MI455X (gfx1250) — compile-verified, matrix operand order checked by reading, not run on hardware
//
#include <hip/hip_runtime.h>
#include <hip/hip_bf16.h>

typedef _Float16 v8h  __attribute__((ext_vector_type(8)));
typedef _Float16 v16h __attribute__((ext_vector_type(16)));
typedef float    v8f  __attribute__((ext_vector_type(8)));

#define EPS_  1e-5f
#define D_    256
#define H1_   64
#define H2_   32
#define O_    8
#define T_    4
#define LDX   264   // 256 + 8 half pad -> conflict-free ds_load_b128
#define LDH   72    // 64 + 8 half pad
#define SENT  0xFFFFFFFFu

// ---------------- prep kernels ----------------

// ws32 layout (u32 words): [0..3]=counts [4..7]=cursor [8..12]=tileSeg prefix
// [16..16+B+64)=perm (sentinel padded). After that: w1 frags (1MB), w2 frags (128KB).
__global__ void k_init(unsigned* ws32, int n) {
  int i = blockIdx.x * blockDim.x + threadIdx.x;
  if (i < 8) ws32[i] = 0u;
  if (i < n) ws32[16 + i] = SENT;
}

__global__ void k_hist(const int* __restrict__ treat, unsigned* ws32, int B) {
  int i = blockIdx.x * blockDim.x + threadIdx.x;
  if (i < B) atomicAdd(&ws32[treat[i] & 3], 1u);
}

__global__ void k_offsets(unsigned* ws32) {
  if (threadIdx.x == 0) {
    unsigned s = 0;
    ws32[8] = 0;
    for (int t = 0; t < 4; ++t) { s += (ws32[t] + 15u) >> 4; ws32[9 + t] = s; }
  }
}

__global__ void k_scatter(const int* __restrict__ treat, unsigned* ws32, int B) {
  int i = blockIdx.x * blockDim.x + threadIdx.x;
  if (i < B) {
    int t = treat[i] & 3;
    unsigned pos = atomicAdd(&ws32[4 + t], 1u);
    ws32[16 + ws32[8 + t] * 16u + pos] = (unsigned)i;
  }
}

// Pack W1/W2 (f32, [o][t][K][n]) into f16 B-fragments matching the documented
// 16-bit operand striping: lane l -> n = l%16; K-group = (l<16)?0:8;
// half e -> K += (e<8) ? e : e+8.  One thread per (fragment, lane).
__global__ void k_wprep(const float* __restrict__ W1, const float* __restrict__ W2,
                        _Float16* __restrict__ w1f, _Float16* __restrict__ w2f) {
  int id = blockIdx.x * blockDim.x + threadIdx.x;
  if (id < 32768) {                 // W1: O*T * 8 ksteps * 4 ntiles = 1024 frags
    int lane = id & 31, f = id >> 5;
    int nt = f & 3, ks = (f >> 2) & 7, tt = (f >> 5) & 3, oo = f >> 7;
    int n  = nt * 16 + (lane & 15);
    int kg = (lane < 16) ? 0 : 8;
    _Float16* dst = w1f + (size_t)id * 16;
    #pragma unroll
    for (int e = 0; e < 16; ++e) {
      int K = ks * 32 + kg + ((e < 8) ? e : e + 8);
      dst[e] = (_Float16)W1[(size_t)((oo * T_ + tt) * D_ + K) * H1_ + n];
    }
  } else if (id < 36864) {          // W2: O*T * 2 ksteps * 2 ntiles = 128 frags
    int id2 = id - 32768;
    int lane = id2 & 31, f = id2 >> 5;
    int nt = f & 1, ks = (f >> 1) & 1, tt = (f >> 2) & 3, oo = f >> 4;
    int n  = nt * 16 + (lane & 15);
    int kg = (lane < 16) ? 0 : 8;
    _Float16* dst = w2f + (size_t)id2 * 16;
    #pragma unroll
    for (int e = 0; e < 16; ++e) {
      int K = ks * 32 + kg + ((e < 8) ? e : e + 8);
      dst[e] = (_Float16)W2[(size_t)((oo * T_ + tt) * H1_ + K) * H2_ + n];
    }
  }
}

// ---------------- main fused kernel ----------------
// One block = one 16-row single-treatment tile; 8 waves, wave o = outcome.
__global__ __launch_bounds__(256) void k_main(
    const float* __restrict__ x,
    const float* __restrict__ b1, const float* __restrict__ g1,
    const float* __restrict__ be1, const float* __restrict__ m1, const float* __restrict__ v1,
    const float* __restrict__ b2, const float* __restrict__ g2,
    const float* __restrict__ be2, const float* __restrict__ m2, const float* __restrict__ v2,
    const float* __restrict__ W3, const float* __restrict__ b3,
    const _Float16* __restrict__ w1f, const _Float16* __restrict__ w2f,
    const unsigned* __restrict__ ws32, float* __restrict__ out)
{
  __shared__ __align__(16) _Float16 xs[16 * LDX];
  __shared__ __align__(16) _Float16 h1s[8 * 16 * LDH];
  __shared__ unsigned rowsS[16];

  const unsigned* seg = ws32 + 8;
  unsigned g = blockIdx.x;
  if (g >= seg[4]) return;                    // block-uniform exit
  int t = 0;
  while (t < 3 && g >= seg[1 + t]) ++t;       // bucket (treatment) of this tile

  const unsigned* perm = ws32 + 16;
  int tid = threadIdx.x;
  if (tid < 16) rowsS[tid] = perm[g * 16u + tid];
  __syncthreads();

  // cooperative gather of 16 x D f32 rows -> f16 LDS tile
  {
    int rowSlot = tid >> 4;
    int colBase = (tid & 15) * 16;
    unsigned rr = rowsS[rowSlot];
    const float* src = x + (size_t)((rr == SENT) ? 0u : rr) * D_ + colBase;
    _Float16* dst = &xs[rowSlot * LDX + colBase];
    #pragma unroll
    for (int j = 0; j < 4; ++j) {
      float4 v = ((const float4*)src)[j];
      dst[j * 4 + 0] = (_Float16)v.x; dst[j * 4 + 1] = (_Float16)v.y;
      dst[j * 4 + 2] = (_Float16)v.z; dst[j * 4 + 3] = (_Float16)v.w;
    }
  }
  __syncthreads();

  int wave = tid >> 5, lane = tid & 31;
  int o = wave;
  int laneN = lane & 15;
  int kg = (lane < 16) ? 0 : 8;   // K-group for A/B frags; also +8 row offset for C

  // ---- layer 1: [16xD] @ W1[o][t] (DxH1), K=256 -> 8 ksteps, N=64 -> 4 ntiles ----
  v8f acc[4] = { {0,0,0,0,0,0,0,0}, {0,0,0,0,0,0,0,0},
                 {0,0,0,0,0,0,0,0}, {0,0,0,0,0,0,0,0} };
  const _Float16* w1base = w1f + (size_t)(o * T_ + t) * (8 * 4 * 32 * 16);
  #pragma unroll
  for (int ks = 0; ks < 8; ++ks) {
    const _Float16* ap = &xs[laneN * LDX + ks * 32 + kg];
    v8h lo = *(const v8h*)ap;
    v8h hi = *(const v8h*)(ap + 16);
    v16h A = __builtin_shufflevector(lo, hi, 0,1,2,3,4,5,6,7,8,9,10,11,12,13,14,15);
    #pragma unroll
    for (int nt = 0; nt < 4; ++nt) {
      v16h Bm = *(const v16h*)(w1base + (size_t)((ks * 4 + nt) * 32 + lane) * 16);
      acc[nt] = __builtin_amdgcn_wmma_f32_16x16x32_f16(
          false, A, false, Bm, (short)0, acc[nt], false, false);
    }
  }

  // ---- epilogue 1: bias + ReLU + BN(eval) -> f16 h1 tile in per-wave LDS ----
  _Float16* hh = &h1s[wave * 16 * LDH];
  int pb1 = (o * T_ + t) * H1_;
  #pragma unroll
  for (int nt = 0; nt < 4; ++nt) {
    int n = nt * 16 + laneN;
    float sc = g1[pb1 + n] * rsqrtf(v1[pb1 + n] + EPS_);
    float sh = be1[pb1 + n] - m1[pb1 + n] * sc;
    float bb = b1[pb1 + n];
    #pragma unroll
    for (int r = 0; r < 8; ++r) {
      float hv = fmaxf(acc[nt][r] + bb, 0.0f) * sc + sh;
      hh[(r + kg) * LDH + n] = (_Float16)hv;      // lanes 16..31 hold rows M+8
    }
  }

  // ---- layer 2: [16xH1] @ W2[o][t] (H1xH2), K=64 -> 2 ksteps, N=32 -> 2 ntiles ----
  v8f acc2[2] = { {0,0,0,0,0,0,0,0}, {0,0,0,0,0,0,0,0} };
  const _Float16* w2base = w2f + (size_t)(o * T_ + t) * (2 * 2 * 32 * 16);
  #pragma unroll
  for (int ks = 0; ks < 2; ++ks) {
    const _Float16* ap = &hh[laneN * LDH + ks * 32 + kg];
    v8h lo = *(const v8h*)ap;
    v8h hi = *(const v8h*)(ap + 16);
    v16h A = __builtin_shufflevector(lo, hi, 0,1,2,3,4,5,6,7,8,9,10,11,12,13,14,15);
    #pragma unroll
    for (int nt = 0; nt < 2; ++nt) {
      v16h Bm = *(const v16h*)(w2base + (size_t)((ks * 2 + nt) * 32 + lane) * 16);
      acc2[nt] = __builtin_amdgcn_wmma_f32_16x16x32_f16(
          false, A, false, Bm, (short)0, acc2[nt], false, false);
    }
  }

  // ---- epilogue 2 + layer 3 (N=1 dot done as FMA + cross-lane reduce) ----
  int pb2 = (o * T_ + t) * H2_;
  float s[8] = {0, 0, 0, 0, 0, 0, 0, 0};
  #pragma unroll
  for (int nt = 0; nt < 2; ++nt) {
    int n = nt * 16 + laneN;
    float sc = g2[pb2 + n] * rsqrtf(v2[pb2 + n] + EPS_);
    float sh = be2[pb2 + n] - m2[pb2 + n] * sc;
    float bb = b2[pb2 + n];
    float w3 = W3[pb2 + n];
    #pragma unroll
    for (int r = 0; r < 8; ++r) {
      float hv = fmaxf(acc2[nt][r] + bb, 0.0f) * sc + sh;
      s[r] += hv * w3;
    }
  }
  #pragma unroll
  for (int mk = 1; mk <= 8; mk <<= 1)       // reduce within each 16-lane half
    #pragma unroll
    for (int r = 0; r < 8; ++r)
      s[r] += __shfl_xor(s[r], mk, 32);

  float bias3 = b3[o * T_ + t];
  if (laneN == 0) {
    #pragma unroll
    for (int r = 0; r < 8; ++r) {
      unsigned rr = rowsS[r + kg];
      if (rr != SENT) out[(size_t)rr * O_ + o] = s[r] + bias3;
    }
  }
}

// ---------------- launcher ----------------
extern "C" void kernel_launch(void* const* d_in, const int* in_sizes, int n_in,
                              void* d_out, int out_size, void* d_ws, size_t ws_size,
                              hipStream_t stream) {
  (void)n_in; (void)out_size; (void)ws_size;
  const float* feat = (const float*)d_in[0];
  const int*   treat= (const int*)d_in[1];
  const float* W1   = (const float*)d_in[2];
  const float* b1   = (const float*)d_in[3];
  const float* g1   = (const float*)d_in[4];
  const float* be1  = (const float*)d_in[5];
  const float* m1   = (const float*)d_in[6];
  const float* v1   = (const float*)d_in[7];
  const float* W2   = (const float*)d_in[8];
  const float* b2   = (const float*)d_in[9];
  const float* g2   = (const float*)d_in[10];
  const float* be2  = (const float*)d_in[11];
  const float* m2   = (const float*)d_in[12];
  const float* v2   = (const float*)d_in[13];
  const float* W3   = (const float*)d_in[14];
  const float* b3   = (const float*)d_in[15];
  float* out = (float*)d_out;

  int B = in_sizes[0] / D_;
  unsigned* ws32 = (unsigned*)d_ws;
  size_t permWords = (size_t)B + 64;
  _Float16* w1f = (_Float16*)((char*)d_ws + 64 + permWords * 4);   // 32B aligned
  _Float16* w2f = w1f + (size_t)O_ * T_ * 8 * 4 * 32 * 16;

  int initN = B + 64;
  k_init   <<<(initN + 255) / 256, 256, 0, stream>>>(ws32, initN);
  k_hist   <<<(B + 255) / 256,     256, 0, stream>>>(treat, ws32, B);
  k_offsets<<<1, 32, 0, stream>>>(ws32);
  k_scatter<<<(B + 255) / 256,     256, 0, stream>>>(treat, ws32, B);
  k_wprep  <<<36864 / 256,         256, 0, stream>>>(W1, W2, w1f, w2f);
  k_main   <<<B / 16 + 4,          256, 0, stream>>>(feat,
      b1, g1, be1, m1, v1, b2, g2, be2, m2, v2, W3, b3, w1f, w2f, ws32, out);
}
